// MultiHeadAttention_22832046146272
// MI455X (gfx1250) — matrix operand order repaired and checked by reading, not compiled, not run on hardware
//
#include <hip/hip_runtime.h>

// ---------------------------------------------------------------------------
// MI455X (gfx1250) multi-head attention, f16 WMMA with f32 accumulation.
// Pipeline: weight transpose/convert -> QKV projection GEMM -> flash attention
//           -> output projection GEMM.  All matrix math on v_wmma_f32_16x16x32_f16.
// LDS is ping-pong double-buffered in all tiled kernels (1 barrier / k-step).
// Workspace layout (needs 40 MiB):
//   [ 0MiB) Wt_q  f16 [1024][1024]   (N-major, K contiguous)
//   [ 2MiB) Wt_k
//   [ 4MiB) Wt_v
//   [ 6MiB) Wt_o
//   [ 8MiB) Qh    f16 [B=2][H=16][S=2048][64]
//   [16MiB) Kh    f16 [2][16][2048][64]
//   [24MiB) Vt    f16 [2][16][64][2048]   (transposed for P*V B-fragments)
//   [32MiB) ctx   f16 [4096][1024]
// ---------------------------------------------------------------------------

typedef _Float16 h4_t  __attribute__((ext_vector_type(4)));
typedef _Float16 h8_t  __attribute__((ext_vector_type(8)));
typedef _Float16 h16_t __attribute__((ext_vector_type(16)));
typedef float    f8_t  __attribute__((ext_vector_type(8)));

#define SWZ(x, m) __int_as_float(__builtin_amdgcn_ds_swizzle(__float_as_int(x), (((m) << 10) | 0x1F)))

__device__ __forceinline__ float red16_max(float x) {
  x = fmaxf(x, SWZ(x, 1));
  x = fmaxf(x, SWZ(x, 2));
  x = fmaxf(x, SWZ(x, 4));
  x = fmaxf(x, SWZ(x, 8));
  return x;
}
__device__ __forceinline__ float red16_sum(float x) {
  x += SWZ(x, 1);
  x += SWZ(x, 2);
  x += SWZ(x, 4);
  x += SWZ(x, 8);
  return x;
}

union h16u { h16_t v; h8_t h[2]; };

// A-fragment (16x32, 16-bit): lane L<16 -> row M=L, K {0..7,16..23};
// lane 16+L -> row M=L, K {8..15,24..31}.  rowptr -> K=0 of this row.
__device__ __forceinline__ h16_t frag_a(const _Float16* rowptr, int half) {
  h16u u;
  u.h[0] = *(const h8_t*)(rowptr + half * 8);
  u.h[1] = *(const h8_t*)(rowptr + 16 + half * 8);
  return u.v;
}

// B-fragment (32x16, 16-bit) from B^T storage (N-major rows, K contiguous):
// lane L<16 -> col N=L, K {0..7,16..23}; lane 16+L -> col N=L, K {8..15,24..31}.
__device__ __forceinline__ h16_t frag_b(const _Float16* rowptr, int half) {
  h16u u;
  u.h[0] = *(const h8_t*)(rowptr + half * 8);
  u.h[1] = *(const h8_t*)(rowptr + 16 + half * 8);
  return u.v;
}

__device__ __forceinline__ f8_t wmma16(h16_t a, h16_t b, f8_t c) {
  return __builtin_amdgcn_wmma_f32_16x16x32_f16(false, a, false, b, (short)0, c, false, false);
}

// ---------------------------------------------------------------------------
// Kernel 0: W[k][n] f32 -> Wt[n][k] f16  (vectorized 4-wide output)
// ---------------------------------------------------------------------------
__global__ __launch_bounds__(256) void wt_kernel(const float* __restrict__ W,
                                                 _Float16* __restrict__ Wt) {
  int t = blockIdx.x * 256 + threadIdx.x;  // over 1024*256
  int k4 = (t & 255) * 4;
  int n = t >> 8;
  const float* src = W + (size_t)k4 * 1024 + n;
  h4_t hv = {(_Float16)src[0], (_Float16)src[1024], (_Float16)src[2048], (_Float16)src[3072]};
  *(h4_t*)(Wt + (size_t)n * 1024 + k4) = hv;
}

// ---------------------------------------------------------------------------
// Tiled GEMM helpers: 128x128 block tile, BK=32, 256 threads (8 waves).
// ---------------------------------------------------------------------------
#define BM 128
#define BN 128
#define BKK 32
#define LDAB 40   // 32 + 8 pad -> 80B row stride, 16B aligned

__device__ __forceinline__ void stage_a_f32(const float* __restrict__ X, int m0, int k0,
                                            int tid, _Float16* __restrict__ As) {
#pragma unroll
  for (int it = 0; it < 4; ++it) {
    int slot = tid + it * 256;  // 0..1023
    int row = slot >> 3;
    int col = (slot & 7) * 4;
    float4 f = *(const float4*)(X + (size_t)(m0 + row) * 1024 + k0 + col);
    h4_t hv = {(_Float16)f.x, (_Float16)f.y, (_Float16)f.z, (_Float16)f.w};
    *(h4_t*)(&As[row * LDAB + col]) = hv;
  }
}

__device__ __forceinline__ void stage_f16(const _Float16* __restrict__ G, int r0, int k0,
                                          int tid, _Float16* __restrict__ Bs) {
#pragma unroll
  for (int it = 0; it < 4; ++it) {
    int slot = tid + it * 256;
    int row = slot >> 3;
    int col = (slot & 7) * 4;
    *(h4_t*)(&Bs[row * LDAB + col]) =
        *(const h4_t*)(G + (size_t)(r0 + row) * 1024 + k0 + col);
  }
}

// ---------------------------------------------------------------------------
// QKV projection GEMM: X(f32)[4096,1024] x Wt(f16)[N,K] + bias -> head layouts.
// ---------------------------------------------------------------------------
__global__ __launch_bounds__(256) void gemm_proj(
    const float* __restrict__ qx, const float* __restrict__ kx, const float* __restrict__ vx,
    const _Float16* __restrict__ Wtq, const _Float16* __restrict__ Wtk, const _Float16* __restrict__ Wtv,
    const float* __restrict__ bq, const float* __restrict__ bk, const float* __restrict__ bv,
    _Float16* __restrict__ Qh, _Float16* __restrict__ Kh, _Float16* __restrict__ Vt) {
  const int mode = blockIdx.z;  // 0=Q 1=K 2=V
  const float*    X    = (mode == 0) ? qx  : (mode == 1) ? kx  : vx;
  const _Float16* Wt   = (mode == 0) ? Wtq : (mode == 1) ? Wtk : Wtv;
  const float*    bias = (mode == 0) ? bq  : (mode == 1) ? bk  : bv;
  _Float16*       dst  = (mode == 0) ? Qh  : (mode == 1) ? Kh  : Vt;

  __shared__ alignas(16) _Float16 As[2][BM * LDAB];
  __shared__ alignas(16) _Float16 Bs[2][BN * LDAB];

  const int m0 = blockIdx.x * BM;
  const int n0 = blockIdx.y * BN;
  const int tid  = threadIdx.x;
  const int wave = tid >> 5, lane = tid & 31;
  const int half = lane >> 4, lidx = lane & 15;
  const int wm = wave >> 1, wn = wave & 1;

  const f8_t zero8 = {0.f, 0.f, 0.f, 0.f, 0.f, 0.f, 0.f, 0.f};
  f8_t acc[2][4];
#pragma unroll
  for (int i = 0; i < 2; ++i)
#pragma unroll
    for (int j = 0; j < 4; ++j) acc[i][j] = zero8;

  stage_a_f32(X, m0, 0, tid, As[0]);
  stage_f16(Wt, n0, 0, tid, Bs[0]);

  for (int k0 = 0; k0 < 1024; k0 += BKK) {
    const int cur = (k0 >> 5) & 1;
    __syncthreads();
    if (k0 + BKK < 1024) {
      stage_a_f32(X, m0, k0 + BKK, tid, As[cur ^ 1]);
      stage_f16(Wt, n0, k0 + BKK, tid, Bs[cur ^ 1]);
    }

    h16_t af[2], bfr[4];
#pragma unroll
    for (int i = 0; i < 2; ++i)
      af[i] = frag_a(&As[cur][(wm * 32 + i * 16 + lidx) * LDAB], half);
#pragma unroll
    for (int j = 0; j < 4; ++j)
      bfr[j] = frag_b(&Bs[cur][(wn * 64 + j * 16 + lidx) * LDAB], half);

#pragma unroll
    for (int i = 0; i < 2; ++i)
#pragma unroll
      for (int j = 0; j < 4; ++j) acc[i][j] = wmma16(af[i], bfr[j], acc[i][j]);
  }

  // epilogue: bias add + head-split store, branch-free inner loop.
  // Q/K: idx = ((bi*16+h)*2048+s)*64 + d, per-row stride 64.
  // V  : idx = ((bi*16+h)*64+d)*2048 + s, per-row stride 1 (vector store).
#pragma unroll
  for (int i = 0; i < 2; ++i) {
#pragma unroll
    for (int j = 0; j < 4; ++j) {
      int n = n0 + wn * 64 + j * 16 + lidx;
      int hh = n >> 6, d = n & 63;
      float bb = bias[n];
      int mfirst = m0 + wm * 32 + i * 16 + half * 8;  // rows mfirst..mfirst+7, same batch
      int bi = mfirst >> 11, s = mfirst & 2047;
      if (mode != 2) {
        size_t base = ((size_t)(bi * 16 + hh) * 2048 + s) * 64 + d;
#pragma unroll
        for (int r = 0; r < 8; ++r)
          dst[base + (size_t)r * 64] = (_Float16)(acc[i][j][r] + bb);
      } else {
        size_t base = ((size_t)(bi * 16 + hh) * 64 + d) * 2048 + s;
        h8_t pack;
#pragma unroll
        for (int r = 0; r < 8; ++r) pack[r] = (_Float16)(acc[i][j][r] + bb);
        *(h8_t*)(dst + base) = pack;  // s is 8-aligned -> 16B aligned
      }
    }
  }
}

// ---------------------------------------------------------------------------
// Flash attention: block = 4 waves = 64 q rows of one (b,h); 32-key tiles,
// K/V tiles ping-pong double-buffered in LDS.
// ---------------------------------------------------------------------------
__device__ __forceinline__ void stage_kv(const _Float16* __restrict__ K,
                                         const _Float16* __restrict__ V, int kb, int tid,
                                         _Float16* __restrict__ Ks, _Float16* __restrict__ Vs) {
#pragma unroll
  for (int it = 0; it < 4; ++it) {
    int slot = tid + it * 128;  // 0..511
    int row = slot >> 4, col = (slot & 15) * 4;
    *(h4_t*)&Ks[row * 72 + col] = *(const h4_t*)(K + (size_t)(kb + row) * 64 + col);
  }
#pragma unroll
  for (int it = 0; it < 4; ++it) {
    int slot = tid + it * 128;
    int row = slot >> 3, col = (slot & 7) * 4;
    *(h4_t*)&Vs[row * 40 + col] = *(const h4_t*)(V + (size_t)row * 2048 + kb + col);
  }
}

__global__ __launch_bounds__(128) void attn_kernel(
    const _Float16* __restrict__ Qh, const _Float16* __restrict__ Kh,
    const _Float16* __restrict__ Vt, _Float16* __restrict__ ctx) {
  const int bh = blockIdx.y;  // 0..31
  const int b = bh >> 4, h = bh & 15;
  const _Float16* Q = Qh + (size_t)bh * 2048 * 64;
  const _Float16* K = Kh + (size_t)bh * 2048 * 64;
  const _Float16* V = Vt + (size_t)bh * 64 * 2048;

  const int tid  = threadIdx.x;
  const int wave = tid >> 5, lane = tid & 31;
  const int half = lane >> 4, lidx = lane & 15;
  const int qbase = blockIdx.x * 64 + wave * 16;

  __shared__ alignas(16) _Float16 Ks[2][32 * 72];     // 32 keys x 64 d
  __shared__ alignas(16) _Float16 Vs[2][64 * 40];     // 64 d x 32 keys
  __shared__ alignas(16) _Float16 Ps[4][16 * 40];     // per-wave P tile

  const _Float16* qrow = Q + (size_t)(qbase + lidx) * 64;
  h16_t aq0 = frag_a(qrow, half);
  h16_t aq1 = frag_a(qrow + 32, half);

  const f8_t zero8 = {0.f, 0.f, 0.f, 0.f, 0.f, 0.f, 0.f, 0.f};
  f8_t O[4];
#pragma unroll
  for (int dj = 0; dj < 4; ++dj) O[dj] = zero8;
  float mrow[8], lrow[8];
#pragma unroll
  for (int r = 0; r < 8; ++r) { mrow[r] = -1e30f; lrow[r] = 0.f; }

  stage_kv(K, V, 0, tid, Ks[0], Vs[0]);

  for (int kb = 0; kb < 2048; kb += 32) {
    const int cur = (kb >> 5) & 1;
    __syncthreads();
    if (kb + 32 < 2048) stage_kv(K, V, kb + 32, tid, Ks[cur ^ 1], Vs[cur ^ 1]);

    // scores S = Q * K^T  (two 16x16 tiles over 32 keys)
    f8_t sc[2];
#pragma unroll
    for (int jt = 0; jt < 2; ++jt) {
      h16_t bk0 = frag_b(&Ks[cur][(jt * 16 + lidx) * 72], half);       // d 0..31
      h16_t bk1 = frag_b(&Ks[cur][(jt * 16 + lidx) * 72 + 32], half);  // d 32..63
      f8_t z = zero8;
      z = wmma16(aq0, bk0, z);
      z = wmma16(aq1, bk1, z);
      sc[jt] = z;
    }

    // online softmax (rows live in the VGPR index; reduce across 16 lanes)
    float tmax[8], corr[8];
#pragma unroll
    for (int r = 0; r < 8; ++r) {
      float a = sc[0][r] * 0.125f;  // 1/sqrt(64)
      float c = sc[1][r] * 0.125f;
      sc[0][r] = a; sc[1][r] = c;
      tmax[r] = red16_max(fmaxf(a, c));
    }
#pragma unroll
    for (int r = 0; r < 8; ++r) {
      float mn = fmaxf(mrow[r], tmax[r]);
      corr[r] = __expf(mrow[r] - mn);
      mrow[r] = mn;
      float p0 = __expf(sc[0][r] - mn);
      float p1 = __expf(sc[1][r] - mn);
      sc[0][r] = p0; sc[1][r] = p1;
      lrow[r] = lrow[r] * corr[r] + red16_sum(p0 + p1);
    }
#pragma unroll
    for (int dj = 0; dj < 4; ++dj)
#pragma unroll
      for (int r = 0; r < 8; ++r) O[dj][r] *= corr[r];

    // C-layout P -> LDS -> A-layout fragment (per-wave buffer, same-wave order)
    _Float16* pw = Ps[wave];
#pragma unroll
    for (int jt = 0; jt < 2; ++jt)
#pragma unroll
      for (int r = 0; r < 8; ++r)
        pw[(half * 8 + r) * 40 + jt * 16 + lidx] = (_Float16)sc[jt][r];
    h16_t ap = frag_a(&pw[lidx * 40], half);

    // O += P * V
#pragma unroll
    for (int dj = 0; dj < 4; ++dj) {
      h16_t bv = frag_b(&Vs[cur][(dj * 16 + lidx) * 40], half);
      O[dj] = wmma16(ap, bv, O[dj]);
    }
  }

  float inv[8];
#pragma unroll
  for (int r = 0; r < 8; ++r) inv[r] = 1.0f / lrow[r];
#pragma unroll
  for (int dj = 0; dj < 4; ++dj)
#pragma unroll
    for (int r = 0; r < 8; ++r) {
      int sq = qbase + half * 8 + r;
      int d = dj * 16 + lidx;
      ctx[(size_t)(b * 2048 + sq) * 1024 + h * 64 + d] = (_Float16)(O[dj][r] * inv[r]);
    }
}

// ---------------------------------------------------------------------------
// Output projection: ctx(f16) x Wt_o + bo -> out(f32)
// ---------------------------------------------------------------------------
__global__ __launch_bounds__(256) void gemm_out(
    const _Float16* __restrict__ ctx, const _Float16* __restrict__ Wto,
    const float* __restrict__ bo, float* __restrict__ out) {
  __shared__ alignas(16) _Float16 As[2][BM * LDAB];
  __shared__ alignas(16) _Float16 Bs[2][BN * LDAB];

  const int m0 = blockIdx.x * BM;
  const int n0 = blockIdx.y * BN;
  const int tid  = threadIdx.x;
  const int wave = tid >> 5, lane = tid & 31;
  const int half = lane >> 4, lidx = lane & 15;
  const int wm = wave >> 1, wn = wave & 1;

  const f8_t zero8 = {0.f, 0.f, 0.f, 0.f, 0.f, 0.f, 0.f, 0.f};
  f8_t acc[2][4];
#pragma unroll
  for (int i = 0; i < 2; ++i)
#pragma unroll
    for (int j = 0; j < 4; ++j) acc[i][j] = zero8;

  stage_f16(ctx, m0, 0, tid, As[0]);
  stage_f16(Wto, n0, 0, tid, Bs[0]);

  for (int k0 = 0; k0 < 1024; k0 += BKK) {
    const int cur = (k0 >> 5) & 1;
    __syncthreads();
    if (k0 + BKK < 1024) {
      stage_f16(ctx, m0, k0 + BKK, tid, As[cur ^ 1]);
      stage_f16(Wto, n0, k0 + BKK, tid, Bs[cur ^ 1]);
    }

    h16_t af[2], bfr[4];
#pragma unroll
    for (int i = 0; i < 2; ++i)
      af[i] = frag_a(&As[cur][(wm * 32 + i * 16 + lidx) * LDAB], half);
#pragma unroll
    for (int j = 0; j < 4; ++j)
      bfr[j] = frag_b(&Bs[cur][(wn * 64 + j * 16 + lidx) * LDAB], half);

#pragma unroll
    for (int i = 0; i < 2; ++i)
#pragma unroll
      for (int j = 0; j < 4; ++j) acc[i][j] = wmma16(af[i], bfr[j], acc[i][j]);
  }

#pragma unroll
  for (int i = 0; i < 2; ++i) {
#pragma unroll
    for (int j = 0; j < 4; ++j) {
      int n = n0 + wn * 64 + j * 16 + lidx;
      float bb = bo[n];
      int mfirst = m0 + wm * 32 + i * 16 + half * 8;
#pragma unroll
      for (int r = 0; r < 8; ++r)
        out[(size_t)(mfirst + r) * 1024 + n] = acc[i][j][r] + bb;
    }
  }
}

// ---------------------------------------------------------------------------
extern "C" void kernel_launch(void* const* d_in, const int* in_sizes, int n_in,
                              void* d_out, int out_size, void* d_ws, size_t ws_size,
                              hipStream_t stream) {
  (void)in_sizes; (void)n_in; (void)out_size; (void)ws_size;

  const float* q  = (const float*)d_in[0];
  const float* k  = (const float*)d_in[1];
  const float* v  = (const float*)d_in[2];
  const float* Wq = (const float*)d_in[3];
  const float* bq = (const float*)d_in[4];
  const float* Wk = (const float*)d_in[5];
  const float* bk = (const float*)d_in[6];
  const float* Wv = (const float*)d_in[7];
  const float* bv = (const float*)d_in[8];
  const float* Wo = (const float*)d_in[9];
  const float* bo = (const float*)d_in[10];
  float* out = (float*)d_out;

  char* ws = (char*)d_ws;
  const size_t MiB = 1024u * 1024u;
  _Float16* Wtq = (_Float16*)(ws + 0 * MiB);
  _Float16* Wtk = (_Float16*)(ws + 2 * MiB);
  _Float16* Wtv = (_Float16*)(ws + 4 * MiB);
  _Float16* Wto = (_Float16*)(ws + 6 * MiB);
  _Float16* Qh  = (_Float16*)(ws + 8 * MiB);
  _Float16* Kh  = (_Float16*)(ws + 16 * MiB);
  _Float16* Vtp = (_Float16*)(ws + 24 * MiB);
  _Float16* ctx = (_Float16*)(ws + 32 * MiB);

  // 1) weight transpose + f16 convert
  wt_kernel<<<1024, 256, 0, stream>>>(Wq, Wtq);
  wt_kernel<<<1024, 256, 0, stream>>>(Wk, Wtk);
  wt_kernel<<<1024, 256, 0, stream>>>(Wv, Wtv);
  wt_kernel<<<1024, 256, 0, stream>>>(Wo, Wto);

  // 2) Q/K/V projections (grid.z selects tensor)
  gemm_proj<<<dim3(32, 8, 3), 256, 0, stream>>>(q, k, v, Wtq, Wtk, Wtv,
                                                bq, bk, bv, Qh, Kh, Vtp);

  // 3) flash attention: x = q-row group of 64, y = (b,h)
  attn_kernel<<<dim3(32, 32), 128, 0, stream>>>(Qh, Kh, Vtp, ctx);

  // 4) output projection
  gemm_out<<<dim3(32, 8), 256, 0, stream>>>(ctx, Wto, bo, out);
}
